// MultiHeadAttention_62328565399995
// MI455X (gfx1250) — matrix operand order repaired and checked by reading, not compiled, not run on hardware
//
#include <hip/hip_runtime.h>
#include <hip/hip_bf16.h>

// ---------------------------------------------------------------------------
// MHA forward for MI455X (gfx1250, wave32, WMMA bf16 path).
// B=2, N=4096, E=512, H=8, D=64.
// ---------------------------------------------------------------------------

#define BN   2
#define NN   4096
#define EE   512
#define HH   8
#define DD   64

typedef __attribute__((ext_vector_type(16))) __bf16 bf16x16;
typedef __attribute__((ext_vector_type(8)))  float  floatx8;

union BF16Tile { bf16x16 v; unsigned u[8]; uint4 q[2]; };

// ---- bf16 packing (round-to-nearest-even via bit trick) -------------------
__device__ __forceinline__ unsigned bf16pair(float lo, float hi) {
    unsigned a = __builtin_bit_cast(unsigned, lo);
    unsigned b = __builtin_bit_cast(unsigned, hi);
    a = (a + 0x7FFFu + ((a >> 16) & 1u)) >> 16;
    b = (b + 0x7FFFu + ((b >> 16) & 1u)) >> 16;
    return (a & 0xFFFFu) | (b << 16);
}
__device__ __forceinline__ unsigned short bf16one(float x) {
    unsigned a = __builtin_bit_cast(unsigned, x);
    a = (a + 0x7FFFu + ((a >> 16) & 1u)) >> 16;
    return (unsigned short)a;
}

// WMMA 16-bit operand layouts (ISA 7.12.2), per lane (m16 = lane%16, h = lane/16):
//   A 16x32: lane row = m16; elements form two contiguous 8-halfword runs at
//            k = 8h (VGPRs 0-3) and k = 16+8h (VGPRs 4-7).
//   B 32x16: lane col = m16; same k order as A: two 8-halfword runs at
//            k = 8h (VGPRs 0-3) and k = 16+8h (VGPRs 4-7).
// So A-chunk = 2x b128 (16B apart +32B), B-chunk likewise (2x b128, 32B apart).

// ---------------------------------------------------------------------------
// Kernel 1: convert the four E x E fp32 weight matrices to bf16 in workspace.
// ---------------------------------------------------------------------------
__global__ void cvt_w_kernel(const float* __restrict__ Wq, const float* __restrict__ Wk,
                             const float* __restrict__ Wv, const float* __restrict__ Wo,
                             unsigned short* __restrict__ out) {
    const int m = blockIdx.y;
    const float* src = (m == 0) ? Wq : (m == 1) ? Wk : (m == 2) ? Wv : Wo;
    unsigned short* dst = out + (size_t)m * (EE * EE);
    const int i = (blockIdx.x * 256 + threadIdx.x) * 4;
    float4 f = *(const float4*)(src + i);
    *(unsigned*)&dst[i]     = bf16pair(f.x, f.y);
    *(unsigned*)&dst[i + 2] = bf16pair(f.z, f.w);
}

// ---------------------------------------------------------------------------
// Kernel 2/4: GEMM  out[g,e] = sum_k A[g,k] * W[e,k] + bias[e]
// Block 256 (8 waves); wave = 16x64 strip; block = 128x64; grid (64, 8).
// ---------------------------------------------------------------------------
template <bool A_BF16, bool OUT_HEADS>
__global__ void __launch_bounds__(256)
proj_gemm_kernel(const void* __restrict__ Ap,
                 const unsigned short* __restrict__ Wb,
                 const float* __restrict__ bias,
                 void* __restrict__ Out) {
    const int t    = threadIdx.x;
    const int wave = t >> 5;
    const int lane = t & 31;
    const int m16  = lane & 15;
    const int half = lane >> 4;

    const int rowbase = blockIdx.x * 128 + wave * 16;
    const int colbase = blockIdx.y * 64;
    const int g = rowbase + m16;   // A-operand row for this lane

    floatx8 acc[4] = {{}, {}, {}, {}};

    for (int k0 = 0; k0 < EE; k0 += 32) {
        BF16Tile am;
        if (A_BF16) {
            const unsigned short* Arow = (const unsigned short*)Ap + (size_t)g * EE;
            const unsigned short* p0 = &Arow[k0 + (half << 3)];
            am.q[0] = *(const uint4*)p0;        // k = k0+8h .. +7
            am.q[1] = *(const uint4*)(p0 + 16); // k = k0+16+8h .. +7
        } else {
            const float* Arow = (const float*)Ap + (size_t)g * EE;
            float4 f0 = *(const float4*)&Arow[k0 + (half << 3)];
            float4 f1 = *(const float4*)&Arow[k0 + (half << 3) + 4];
            float4 f2 = *(const float4*)&Arow[k0 + 16 + (half << 3)];
            float4 f3 = *(const float4*)&Arow[k0 + 16 + (half << 3) + 4];
            am.u[0] = bf16pair(f0.x, f0.y); am.u[1] = bf16pair(f0.z, f0.w);
            am.u[2] = bf16pair(f1.x, f1.y); am.u[3] = bf16pair(f1.z, f1.w);
            am.u[4] = bf16pair(f2.x, f2.y); am.u[5] = bf16pair(f2.z, f2.w);
            am.u[6] = bf16pair(f3.x, f3.y); am.u[7] = bf16pair(f3.z, f3.w);
        }
#pragma unroll
        for (int nt = 0; nt < 4; ++nt) {
            BF16Tile bm;
            const unsigned short* Brow =
                &Wb[(size_t)(colbase + nt * 16 + m16) * EE + k0 + (half << 3)];
            bm.q[0] = *(const uint4*)Brow;
            bm.q[1] = *(const uint4*)(Brow + 16);
            acc[nt] = __builtin_amdgcn_wmma_f32_16x16x32_bf16(
                false, am.v, false, bm.v, (short)0, acc[nt], false, false);
        }
    }

#pragma unroll
    for (int nt = 0; nt < 4; ++nt) {
        const int e  = colbase + nt * 16 + m16;        // C/D column = lane%16
        const float be = bias[e];
#pragma unroll
        for (int r = 0; r < 8; ++r) {
            const float o  = acc[nt][r] + be;
            const int  row = rowbase + r + (half << 3); // C/D row = r + 8*half
            if (OUT_HEADS) {
                const int b = row >> 12, n = row & (NN - 1);
                const int h = e >> 6,   d = e & (DD - 1);
                ((unsigned short*)Out)[((((size_t)b * HH + h) * NN + n) << 6) + d] = bf16one(o);
            } else {
                ((float*)Out)[(size_t)row * EE + e] = o;
            }
        }
    }
}

// ---------------------------------------------------------------------------
// Kernel 3: flash attention over one (b,h). Q,K,V bf16 [N x 64].
// Block 256 (8 waves), 128 query rows/block, 16 per wave.
// Double-buffered 64-key LDS tiles, register-staged software pipeline:
//   sync -> issue next-tile global loads -> WMMA/softmax on current tile ->
//   dump staged regs into other LDS buffer -> loop.
// Row pad = 72 halves (144B): every WMMA operand run is 16B aligned -> b128
// DS loads with immediate offsets; bank-group permutation (9m mod 16) is
// conflict-free across a 16-lane half.
// ---------------------------------------------------------------------------
#define TSTR 72
#define TILE (64 * TSTR)

__global__ void __launch_bounds__(256, 1)
flash_attn_kernel(const unsigned short* __restrict__ Qp,
                  const unsigned short* __restrict__ Kp,
                  const unsigned short* __restrict__ Vp,
                  unsigned short* __restrict__ At) {
    __shared__ __align__(16) unsigned short sK [2 * TILE];       // [buf][key][d]
    __shared__ __align__(16) unsigned short sVt[2 * TILE];       // [buf][d][key]
    __shared__ __align__(16) unsigned short sP [8 * 16 * TSTR];  // per-wave P

    const int t    = threadIdx.x;
    const int wave = t >> 5;
    const int lane = t & 31;
    const int m16  = lane & 15;
    const int half = lane >> 4;

    const int bh = blockIdx.y;
    const int b  = bh >> 3, h = bh & 7;
    const size_t base = (size_t)bh * NN * DD;

    const float SCL2 = 0.125f * 1.4426950408889634f;  // (1/sqrt(D)) * log2(e)

    // --- this wave's 16x64 Q strip, loaded once in A-operand layout --------
    BF16Tile qa[2];
    {
        const unsigned short* qrow =
            Qp + base + (size_t)(blockIdx.x * 128 + wave * 16 + m16) * DD;
#pragma unroll
        for (int c = 0; c < 2; ++c) {
            const unsigned short* p0 = qrow + c * 32 + (half << 3);
            qa[c].q[0] = *(const uint4*)p0;
            qa[c].q[1] = *(const uint4*)(p0 + 16);
        }
    }

    floatx8 Oacc[4] = {{}, {}, {}, {}};
    float mrun[8], lrun[8];
#pragma unroll
    for (int r = 0; r < 8; ++r) { mrun[r] = -1e30f; lrun[r] = 0.0f; }

    unsigned short* sPw = sP + wave * (16 * TSTR);

    // --- register staging helpers (each thread moves 4x8B of K and of V) ---
    uint2 kx[4], vx[4];
    auto stage_load = [&](int n0) {
#pragma unroll
        for (int p = 0; p < 4; ++p) {
            const int idx = t + p * 256;
            const int r   = idx >> 4;
            const int c   = (idx & 15) << 2;
            kx[p] = *(const uint2*)&Kp[base + (size_t)(n0 + r) * DD + c];
            vx[p] = *(const uint2*)&Vp[base + (size_t)(n0 + r) * DD + c];
        }
    };
    auto stage_store = [&](int buf) {
        unsigned short* dK = sK  + buf * TILE;
        unsigned short* dV = sVt + buf * TILE;
#pragma unroll
        for (int p = 0; p < 4; ++p) {
            const int idx = t + p * 256;
            const int r   = idx >> 4;
            const int c   = (idx & 15) << 2;
            *(uint2*)&dK[r * TSTR + c] = kx[p];
            dV[(c + 0) * TSTR + r] = (unsigned short)(kx[p].x & 0u) | (unsigned short)(vx[p].x & 0xFFFFu);
            dV[(c + 1) * TSTR + r] = (unsigned short)(vx[p].x >> 16);
            dV[(c + 2) * TSTR + r] = (unsigned short)(vx[p].y & 0xFFFFu);
            dV[(c + 3) * TSTR + r] = (unsigned short)(vx[p].y >> 16);
        }
    };

    stage_load(0);
    stage_store(0);

    for (int it = 0; it < NN / 64; ++it) {
        __syncthreads();  // current buffer staged; other buffer's readers done

        if (it + 1 < NN / 64) stage_load((it + 1) * 64);  // hide global latency

        const unsigned short* curK = sK  + (it & 1) * TILE;
        const unsigned short* curV = sVt + (it & 1) * TILE;

        // ---- S = Q * K^T : four 16x16 tiles, two K=32 chunks each ---------
        floatx8 sacc[4];
#pragma unroll
        for (int nt = 0; nt < 4; ++nt) {
            floatx8 a = {};
#pragma unroll
            for (int c = 0; c < 2; ++c) {
                BF16Tile bm;
                const unsigned short* pb =
                    &curK[(nt * 16 + m16) * TSTR + c * 32 + (half << 3)];
                bm.q[0] = *(const uint4*)pb;
                bm.q[1] = *(const uint4*)(pb + 16);
                a = __builtin_amdgcn_wmma_f32_16x16x32_bf16(
                    false, qa[c].v, false, bm.v, (short)0, a, false, false);
            }
            sacc[nt] = a;
        }

        // ---- online softmax (row = r + 8*half; 16 lanes share a row) ------
        float mnew[8], alpha[8], rsum[8];
#pragma unroll
        for (int r = 0; r < 8; ++r) {
            float mx = fmaxf(fmaxf(sacc[0][r], sacc[1][r]), fmaxf(sacc[2][r], sacc[3][r]));
            mx = fmaxf(mx, __shfl_xor(mx, 1, 16));
            mx = fmaxf(mx, __shfl_xor(mx, 2, 16));
            mx = fmaxf(mx, __shfl_xor(mx, 4, 16));
            mx = fmaxf(mx, __shfl_xor(mx, 8, 16));
            const float mn = fmaxf(mrun[r], mx);
            alpha[r] = __builtin_amdgcn_exp2f((mrun[r] - mn) * SCL2);
            mnew[r]  = mn;
            mrun[r]  = mn;
            rsum[r]  = 0.0f;
        }
#pragma unroll
        for (int nt = 0; nt < 4; ++nt) {
#pragma unroll
            for (int r = 0; r < 8; ++r) {
                const float pv = __builtin_amdgcn_exp2f((sacc[nt][r] - mnew[r]) * SCL2);
                rsum[r] += pv;
                sPw[(r + (half << 3)) * TSTR + nt * 16 + m16] = bf16one(pv);
            }
        }
#pragma unroll
        for (int r = 0; r < 8; ++r) {
            float s = rsum[r];
            s += __shfl_xor(s, 1, 16);
            s += __shfl_xor(s, 2, 16);
            s += __shfl_xor(s, 4, 16);
            s += __shfl_xor(s, 8, 16);
            lrun[r] = lrun[r] * alpha[r] + s;
#pragma unroll
            for (int dt = 0; dt < 4; ++dt) Oacc[dt][r] *= alpha[r];
        }
        // sP write->read is same-wave DS traffic: program order, no barrier.

        // ---- O += P * V -----------------------------------------------------
        BF16Tile pa[2];
#pragma unroll
        for (int c = 0; c < 2; ++c) {
            const unsigned short* pp = &sPw[m16 * TSTR + c * 32 + (half << 3)];
            pa[c].q[0] = *(const uint4*)pp;
            pa[c].q[1] = *(const uint4*)(pp + 16);
        }
#pragma unroll
        for (int dt = 0; dt < 4; ++dt) {
#pragma unroll
            for (int c = 0; c < 2; ++c) {
                BF16Tile bv;
                const unsigned short* pb =
                    &curV[(dt * 16 + m16) * TSTR + c * 32 + (half << 3)];
                bv.q[0] = *(const uint4*)pb;
                bv.q[1] = *(const uint4*)(pb + 16);
                Oacc[dt] = __builtin_amdgcn_wmma_f32_16x16x32_bf16(
                    false, pa[c].v, false, bv.v, (short)0, Oacc[dt], false, false);
            }
        }

        if (it + 1 < NN / 64) stage_store((it + 1) & 1);  // fill other buffer
    }

    // ---- epilogue: normalize, store bf16 row-major [B,N,E] ----------------
    float inv[8];
#pragma unroll
    for (int r = 0; r < 8; ++r) inv[r] = 1.0f / lrun[r];
#pragma unroll
    for (int dt = 0; dt < 4; ++dt)
#pragma unroll
        for (int r = 0; r < 8; ++r) {
            const float o  = Oacc[dt][r] * inv[r];
            const int qrow = blockIdx.x * 128 + wave * 16 + r + (half << 3);
            const int d    = dt * 16 + m16;
            At[((size_t)b * NN + qrow) * EE + h * DD + d] = bf16one(o);
        }
}

// ---------------------------------------------------------------------------
// Launch: cvt weights -> QKV projections -> flash attention -> out projection
// Workspace map (bytes): [0,2M) W bf16 x4 | [2M,10M) Qp | [10M,18M) Kp |
//                        [18M,26M) Vp | [26M,34M) attn bf16
// ---------------------------------------------------------------------------
extern "C" void kernel_launch(void* const* d_in, const int* in_sizes, int n_in,
                              void* d_out, int out_size, void* d_ws, size_t ws_size,
                              hipStream_t stream) {
    (void)in_sizes; (void)n_in; (void)out_size; (void)ws_size;
    const float* q  = (const float*)d_in[0];
    const float* k  = (const float*)d_in[1];
    const float* v  = (const float*)d_in[2];
    const float* Wq = (const float*)d_in[3];
    const float* bq = (const float*)d_in[4];
    const float* Wk = (const float*)d_in[5];
    const float* bk = (const float*)d_in[6];
    const float* Wv = (const float*)d_in[7];
    const float* bv = (const float*)d_in[8];
    const float* Wo = (const float*)d_in[9];
    const float* bo = (const float*)d_in[10];

    char* ws = (char*)d_ws;
    unsigned short* wbf = (unsigned short*)ws;                          // 4 x 512KB
    unsigned short* Qp  = (unsigned short*)(ws + (size_t)2  * (1 << 20));
    unsigned short* Kp  = (unsigned short*)(ws + (size_t)10 * (1 << 20));
    unsigned short* Vp  = (unsigned short*)(ws + (size_t)18 * (1 << 20));
    unsigned short* At  = (unsigned short*)(ws + (size_t)26 * (1 << 20));

    cvt_w_kernel<<<dim3(EE * EE / 1024, 4), 256, 0, stream>>>(Wq, Wk, Wv, Wo, wbf);

    proj_gemm_kernel<false, true><<<dim3(64, 8), 256, 0, stream>>>(q, wbf,               bq, Qp);
    proj_gemm_kernel<false, true><<<dim3(64, 8), 256, 0, stream>>>(k, wbf + 1 * EE * EE, bk, Kp);
    proj_gemm_kernel<false, true><<<dim3(64, 8), 256, 0, stream>>>(v, wbf + 2 * EE * EE, bv, Vp);

    flash_attn_kernel<<<dim3(NN / 128, BN * HH), 256, 0, stream>>>(Qp, Kp, Vp, At);

    proj_gemm_kernel<true, false><<<dim3(64, 8), 256, 0, stream>>>(At, wbf + 3 * EE * EE, bo, (float*)d_out);
}
